// BidirectionalEdgeGraphNetwork_70420283785781
// MI455X (gfx1250) — matrix operand order repaired and checked by reading, not compiled, not run on hardware
//
#include <hip/hip_runtime.h>

// ---------------------------------------------------------------------------
// BidirectionalEdgeGraphNetwork on MI455X (gfx1250, wave32, WMMA).
// All GEMM operands in bf16 in HBM; LDS staging via double-buffered
// global_load_async_to_lds_b128 (ASYNCcnt software pipeline), feeding
// v_wmma_f32_16x16x32_bf16 with f32 accumulation.
// ---------------------------------------------------------------------------

#define GN 2048
#define GE 32768
#define DDN 512
#define DDE 512
#define NHD 8
#define HDM 64
#define NLY 2

typedef unsigned short u16;
typedef __bf16 bf16;
typedef bf16 v16bf __attribute__((ext_vector_type(16)));
typedef float v8f __attribute__((ext_vector_type(8)));
typedef int gv4i __attribute__((vector_size(16)));  // matches builtin's V4i

#define LDSK 40  // padded LDS row stride in bf16 elements (32 data + 8 pad)

#if __has_builtin(__builtin_amdgcn_global_load_async_to_lds_b128)
#define ASYNC_OK 1
#endif

__device__ __forceinline__ u16 f2bf(float f) {
  unsigned u = __float_as_uint(f);
  unsigned r = u + 0x7FFFu + ((u >> 16) & 1u);  // round-to-nearest-even
  return (u16)(r >> 16);
}

// 16-byte global->LDS copy: async-to-LDS when the toolchain exposes it.
__device__ __forceinline__ void cp16(const u16* g, u16* l) {
#if defined(ASYNC_OK)
  __builtin_amdgcn_global_load_async_to_lds_b128(
      (__attribute__((address_space(1))) gv4i*)g,
      (__attribute__((address_space(3))) gv4i*)l, 0, 0);
#else
  *(uint4*)l = *(const uint4*)g;
#endif
}
// Wait until at most N async copies remain in flight (partial wait keeps the
// next tile's copies overlapped with compute).
template <int N>
__device__ __forceinline__ void cp_wait_n() {
#if defined(ASYNC_OK)
#if __has_builtin(__builtin_amdgcn_s_wait_asynccnt)
  __builtin_amdgcn_s_wait_asynccnt(N);
#else
  asm volatile("s_wait_asynccnt %0" ::"i"(N) : "memory");
#endif
#endif
}

struct GemmEpi {
  const float* bias;   // epi 0/2
  const float* svec;   // epi 2: per-col scale vector (etg)
  const float* shvec;  // epi 2: per-col shift (etbb)
  const float* maskp;  // epi 3: additive mask matrix (f32)
  float alpha;         // epi 3
  float sconst;        // epi 2 (bn_scale)
  int relu;
  int epi;             // 0: bias(+relu)  2: (acc+b)*s+sh, relu  3: acc*alpha+mask
  int maskLd;
};

struct GatherArgs { const int* src; const int* dst; const int* rev;
                    const u16* nf; const u16* ef; const u16* zeros; };
struct ConcatArgs { const u16* a0; const u16* s1; };

// A fragment: 16x32 bf16, lane m=l&15, k-groups (l>>4)*8 and 16+(l>>4)*8
__device__ __forceinline__ v16bf load_fragA(const u16* ls, int row0, int lane) {
  int m = row0 + (lane & 15);
  int g = (lane >> 4) << 3;
  const u16* p = ls + m * LDSK;
  union { v16bf v; uint4 q[2]; } u;
  u.q[0] = *(const uint4*)(p + g);
  u.q[1] = *(const uint4*)(p + 16 + g);
  return u.v;
}
// B fragment: 32x16 bf16, lane n=l&15, k-groups (l>>4)*8 and 16+(l>>4)*8
__device__ __forceinline__ v16bf load_fragB(const u16* ls, int col0, int lane) {
  int n = col0 + (lane & 15);
  int kb = (lane >> 4) << 3;
  const u16* p = ls + n * LDSK + kb;
  union { v16bf v; uint4 q[2]; } u;
  u.q[0] = *(const uint4*)p;
  u.q[1] = *(const uint4*)(p + 16);
  return u.v;
}

// Tiled GEMM: C[M x Nc] = epi( A[M x K] * B^T ), A/B bf16, acc f32.
// AMODE: 0 plain A, 1 gather edge rows [nf[src]|ef|ef[rev]|nf[dst]], 2 concat [a0|s1]
// BTRANS: 0 -> B is [Nc x K] (weights); 1 -> B is [K x Nc] (PV GEMM)
// outbf: 1 -> store bf16 (u16), 0 -> store f32
template <int AMODE, int BTRANS>
__global__ __launch_bounds__(256) void gemm_kernel(
    int M, int Nc, int K,
    const u16* A, int lda, long long aHS,
    const u16* B, int ldb, long long bHS,
    void* Cv, int ldc, long long cHS, int outbf,
    GatherArgs ga, ConcatArgs ca, GemmEpi ep) {
  __shared__ __align__(16) u16 lsA[2][128 * LDSK];
  __shared__ __align__(16) u16 lsB[2][64 * LDSK];

  const int tid = threadIdx.x;
  const int lane = tid & 31;
  const int wave = tid >> 5;  // 8 waves (wave32)
  const int wm = wave >> 1;
  const int wn = wave & 1;
  const int m0 = blockIdx.y * 128;
  const int n0 = blockIdx.x * 64;
  const int z = blockIdx.z;
  if (AMODE == 0) A += (size_t)z * (size_t)aHS;
  B += (size_t)z * (size_t)bHS;
  const size_t zC = (size_t)z * (size_t)cHS;

  // per-thread async issues per stage: 2 (A) + 1 (B, weight path only)
  constexpr int STAGE_ASYNC = (BTRANS == 0) ? 3 : 2;

  auto stage = [&](int buf, int k0) {
    // ---- A tile: 128 rows x 32 bf16, 16B chunks ---------------------------
#pragma unroll
    for (int it = 0; it < 2; ++it) {
      int t = tid + it * 256;  // 0..511
      int row = t >> 2;        // 0..127
      int c16 = (t & 3) << 3;  // 0,8,16,24 (bf16 elements)
      const u16* gp;
      if (AMODE == 0) {
        gp = A + (size_t)(m0 + row) * lda + k0 + c16;
      } else if (AMODE == 1) {
        int e = m0 + row;
        int seg = k0 >> 9;
        int within = (k0 & 511) + c16;
        const u16* base;
        if (seg == 0)      base = ga.nf + (size_t)ga.src[e] * DDN;
        else if (seg == 1) base = ga.ef + (size_t)e * DDE;
        else if (seg == 2) { int r = ga.rev[e];
                             base = (r >= 0) ? (ga.ef + (size_t)r * DDE) : ga.zeros; }
        else               base = ga.nf + (size_t)ga.dst[e] * DDN;
        gp = base + within;
      } else {  // AMODE == 2, K = 1024
        int seg = k0 >> 9;
        int within = (k0 & 511) + c16;
        const u16* base = (seg == 0) ? ca.a0 : ca.s1;
        gp = base + (size_t)(m0 + row) * DDN + within;
      }
      cp16(gp, &lsA[buf][row * LDSK + c16]);
    }
    // ---- B tile: 64 x 32 --------------------------------------------------
    if (BTRANS == 0) {
      int n = tid >> 2;        // 0..63
      int c16 = (tid & 3) << 3;
      cp16(B + (size_t)(n0 + n) * ldb + k0 + c16, &lsB[buf][n * LDSK + c16]);
    } else {
      // transpose-stage: memory is [K x Nc]; read 8 contiguous n, scatter k
      int k = tid >> 3;        // 0..31
      int n8 = (tid & 7) << 3; // 0..56
      union { uint4 q; u16 h[8]; } u;
      u.q = *(const uint4*)(B + (size_t)(k0 + k) * ldb + n0 + n8);
#pragma unroll
      for (int j = 0; j < 8; ++j) lsB[buf][(n8 + j) * LDSK + k] = u.h[j];
    }
  };

  v8f acc[2][2];
#pragma unroll
  for (int i = 0; i < 2; ++i)
#pragma unroll
    for (int j = 0; j < 2; ++j)
#pragma unroll
      for (int e = 0; e < 8; ++e) acc[i][j][e] = 0.0f;

  stage(0, 0);  // prologue: tile 0 in flight

  int cur = 0;
  for (int k0 = 0; k0 < K; k0 += 32, cur ^= 1) {
    if (k0 + 32 < K) {
      stage(cur ^ 1, k0 + 32);   // keep next tile's copies in flight
      cp_wait_n<STAGE_ASYNC>();  // wait only for the *current* tile
    } else {
      cp_wait_n<0>();
    }
    __syncthreads();

    v16bf af0 = load_fragA(lsA[cur], wm * 32, lane);
    v16bf af1 = load_fragA(lsA[cur], wm * 32 + 16, lane);
    v16bf bf0 = load_fragB(lsB[cur], wn * 32, lane);
    v16bf bf1 = load_fragB(lsB[cur], wn * 32 + 16, lane);
    acc[0][0] = __builtin_amdgcn_wmma_f32_16x16x32_bf16(false, af0, false, bf0, (short)0, acc[0][0], false, false);
    acc[0][1] = __builtin_amdgcn_wmma_f32_16x16x32_bf16(false, af0, false, bf1, (short)0, acc[0][1], false, false);
    acc[1][0] = __builtin_amdgcn_wmma_f32_16x16x32_bf16(false, af1, false, bf0, (short)0, acc[1][0], false, false);
    acc[1][1] = __builtin_amdgcn_wmma_f32_16x16x32_bf16(false, af1, false, bf1, (short)0, acc[1][1], false, false);
    __syncthreads();  // all reads of buf `cur` done before it is restaged
  }

  // ---- epilogue: acc vgpr r -> row r + 8*(lane>>4), col lane&15 -----------
  int rowAdd = (lane >> 4) << 3;
#pragma unroll
  for (int i = 0; i < 2; ++i)
#pragma unroll
    for (int j = 0; j < 2; ++j) {
      int baseRow = m0 + wm * 32 + i * 16 + rowAdd;
      int col = n0 + wn * 32 + j * 16 + (lane & 15);
#pragma unroll
      for (int r = 0; r < 8; ++r) {
        int row = baseRow + r;
        float v = acc[i][j][r];
        if (ep.epi == 3) {
          v = v * ep.alpha + ep.maskp[(size_t)row * ep.maskLd + col];
        } else {
          if (ep.bias) v += ep.bias[col];
          if (ep.epi == 2) v = v * (ep.sconst * ep.svec[col]) + ep.shvec[col];
          if (ep.relu) v = fmaxf(v, 0.0f);
        }
        size_t idx = zC + (size_t)row * ldc + col;
        if (outbf) ((u16*)Cv)[idx] = f2bf(v);
        else       ((float*)Cv)[idx] = v;
      }
    }
}

// ---------------------------------------------------------------------------
__global__ __launch_bounds__(256) void cvt_bf16(const float* in, u16* out, size_t n) {
  size_t i = ((size_t)blockIdx.x * 256 + threadIdx.x) * 8;
  if (i >= n) return;
  float4 a = *(const float4*)(in + i);
  float4 b = *(const float4*)(in + i + 4);
  *(ushort4*)(out + i)     = make_ushort4(f2bf(a.x), f2bf(a.y), f2bf(a.z), f2bf(a.w));
  *(ushort4*)(out + i + 4) = make_ushort4(f2bf(b.x), f2bf(b.y), f2bf(b.z), f2bf(b.w));
}
__global__ __launch_bounds__(256) void sum_cvt_bf16(const float* a, const float* b, u16* out,
                                                    size_t n) {
  size_t i = ((size_t)blockIdx.x * 256 + threadIdx.x) * 8;
  if (i >= n) return;
  float4 a0 = *(const float4*)(a + i), a1 = *(const float4*)(a + i + 4);
  float4 b0 = *(const float4*)(b + i), b1 = *(const float4*)(b + i + 4);
  *(ushort4*)(out + i) = make_ushort4(f2bf(a0.x + b0.x), f2bf(a0.y + b0.y),
                                      f2bf(a0.z + b0.z), f2bf(a0.w + b0.w));
  *(ushort4*)(out + i + 4) = make_ushort4(f2bf(a1.x + b1.x), f2bf(a1.y + b1.y),
                                          f2bf(a1.z + b1.z), f2bf(a1.w + b1.w));
}
__global__ __launch_bounds__(256) void fill_f32(float* p, float v, size_t n) {
  size_t i = (size_t)blockIdx.x * 256 + threadIdx.x;
  if (i < n) p[i] = v;
}
__global__ __launch_bounds__(256) void fill_i32(int* p, int v, size_t n) {
  size_t i = (size_t)blockIdx.x * 256 + threadIdx.x;
  if (i < n) p[i] = v;
}
__global__ __launch_bounds__(256) void mask_init(float* mask) {
  size_t i = (size_t)blockIdx.x * 256 + threadIdx.x;
  if (i >= (size_t)GN * GN) return;
  int r = (int)(i / GN), c = (int)(i % GN);
  mask[i] = (r == c) ? 1.0f : -__builtin_inff();
}
__global__ __launch_bounds__(256) void mask_edges(float* mask, const int* src, const int* dst,
                                                  const float* pos) {
  int e = blockIdx.x * 256 + threadIdx.x;
  if (e >= GE) return;
  int s = src[e], d = dst[e];
  float dx = pos[s * 3 + 0] - pos[d * 3 + 0];
  float dy = pos[s * 3 + 1] - pos[d * 3 + 1];
  float dz = pos[s * 3 + 2] - pos[d * 3 + 2];
  float w = 1.0f / (sqrtf(dx * dx + dy * dy + dz * dz) + 1e-6f);
  mask[(size_t)s * GN + d] = w;
  mask[(size_t)d * GN + s] = w;
}
__global__ __launch_bounds__(256) void rev_set(int* table, const int* src, const int* dst) {
  int e = blockIdx.x * 256 + threadIdx.x;
  if (e >= GE) return;
  table[(size_t)src[e] * GN + dst[e]] = e;
}
__global__ __launch_bounds__(256) void rev_get(const int* table, const int* src, const int* dst,
                                               int* rev) {
  int e = blockIdx.x * 256 + threadIdx.x;
  if (e >= GE) return;
  rev[e] = table[(size_t)dst[e] * GN + src[e]];  // -1 if reverse edge absent
}
// f32 numerics; emits bf16 probabilities for the PV WMMA GEMM.
__global__ __launch_bounds__(256) void softmax_rows(float* S, u16* P, int ncols) {
  float* p = S + (size_t)blockIdx.x * ncols;
  u16* q = P + (size_t)blockIdx.x * ncols;
  int tid = threadIdx.x;
  __shared__ float red[8];
  float m = -__builtin_inff();
  for (int j = tid; j < ncols; j += 256) m = fmaxf(m, p[j]);
  for (int o = 16; o; o >>= 1) m = fmaxf(m, __shfl_xor(m, o, 32));
  if ((tid & 31) == 0) red[tid >> 5] = m;
  __syncthreads();
  m = red[0];
#pragma unroll
  for (int i = 1; i < 8; ++i) m = fmaxf(m, red[i]);
  float s = 0.0f;
  for (int j = tid; j < ncols; j += 256) {
    float e = __expf(p[j] - m);
    p[j] = e;
    s += e;
  }
  for (int o = 16; o; o >>= 1) s += __shfl_xor(s, o, 32);
  __syncthreads();
  if ((tid & 31) == 0) red[tid >> 5] = s;
  __syncthreads();
  s = 0.0f;
#pragma unroll
  for (int i = 0; i < 8; ++i) s += red[i];
  float inv = 1.0f / s;
  for (int j = tid; j < ncols; j += 256) q[j] = f2bf(p[j] * inv);
}
// te >= 0 (post-ReLU): uint bit compare == float compare; init = 0.0f
__global__ __launch_bounds__(256) void scatter_max(const float* te, const int* src, const int* dst,
                                                   float* subj, float* obj) {
  size_t i = (size_t)blockIdx.x * 256 + threadIdx.x;
  if (i >= (size_t)GE * DDE) return;
  int e = (int)(i >> 9);
  int j = (int)(i & 511);
  unsigned bits = __float_as_uint(te[i]);
  atomicMax((unsigned*)&subj[(size_t)src[e] * DDE + j], bits);
  atomicMax((unsigned*)&obj[(size_t)dst[e] * DDE + j], bits);
}

// ---------------------------------------------------------------------------
extern "C" void kernel_launch(void* const* d_in, const int* in_sizes, int n_in,
                              void* d_out, int out_size, void* d_ws, size_t ws_size,
                              hipStream_t stream) {
  const float* x     = (const float*)d_in[0];
  const float* ef_in = (const float*)d_in[1];
  const float* pos   = (const float*)d_in[2];
  const float* inw   = (const float*)d_in[3];
  const float* inb   = (const float*)d_in[4];
  const float* ow    = (const float*)d_in[5];
  const float* ob    = (const float*)d_in[6];
  const float* etw   = (const float*)d_in[7];
  const float* etb   = (const float*)d_in[8];
  const float* etg   = (const float*)d_in[9];
  const float* etbb  = (const float*)d_in[10];
  const float* unw   = (const float*)d_in[11];
  const float* unb   = (const float*)d_in[12];
  const float* uew   = (const float*)d_in[13];
  const float* ueb   = (const float*)d_in[14];
  const int*   ei    = (const int*)d_in[15];
  const int* srcI = ei;
  const int* dstI = ei + GE;

  char* w = (char*)d_ws;
  auto carve = [&](size_t bytes) -> char* {
    char* p = w;
    w += (bytes + 255) & ~(size_t)255;
    return p;
  };
  float* mask    = (float*)carve((size_t)GN * GN * 4);
  int*   table   = (int*)carve((size_t)GN * GN * 4);
  int*   rev     = (int*)carve((size_t)GE * 4);
  u16*   nfA_bf  = (u16*)carve((size_t)GN * DDN * 2);
  u16*   nfB_bf  = (u16*)carve((size_t)GN * DDN * 2);
  u16*   efA_bf  = (u16*)carve((size_t)GE * DDE * 2);
  u16*   efB_bf  = (u16*)carve((size_t)GE * DDE * 2);
  u16*   qkv_bf  = (u16*)carve((size_t)GN * 3 * DDN * 2);
  float* scores  = (float*)carve((size_t)NHD * GN * GN * 4);
  u16*   probs   = (u16*)carve((size_t)NHD * GN * GN * 2);
  u16*   ctx_bf  = (u16*)carve((size_t)GN * DDN * 2);
  u16*   xupd_bf = (u16*)carve((size_t)GN * DDN * 2);
  float* te      = (float*)carve((size_t)GE * DDE * 4);
  float* subj    = (float*)carve((size_t)GN * DDE * 4);
  float* obj     = (float*)carve((size_t)GN * DDE * 4);
  u16*   s1_bf   = (u16*)carve((size_t)GN * DDE * 2);
  u16*   zeros   = (u16*)carve((size_t)DDE * 2);
  u16*   inw_bf  = (u16*)carve((size_t)NLY * 3 * DDN * DDN * 2);
  u16*   ow_bf   = (u16*)carve((size_t)NLY * DDN * DDN * 2);
  u16*   etw_bf  = (u16*)carve((size_t)NLY * DDE * DDE * 2);
  u16*   unw_bf  = (u16*)carve((size_t)NLY * DDN * (DDN + DDE) * 2);
  u16*   uew_bf  = (u16*)carve((size_t)NLY * DDE * (2 * DDN + 2 * DDE) * 2);

  auto cvt = [&](const float* in, u16* out, size_t n) {
    cvt_bf16<<<dim3((unsigned)(n / 8 / 256)), 256, 0, stream>>>(in, out, n);
  };
  cvt(x, nfA_bf, (size_t)GN * DDN);
  cvt(ef_in, efA_bf, (size_t)GE * DDE);
  cvt(inw, inw_bf, (size_t)NLY * 3 * DDN * DDN);
  cvt(ow, ow_bf, (size_t)NLY * DDN * DDN);
  cvt(etw, etw_bf, (size_t)NLY * DDE * DDE);
  cvt(unw, unw_bf, (size_t)NLY * DDN * (DDN + DDE));
  cvt(uew, uew_bf, (size_t)NLY * DDE * (2 * DDN + 2 * DDE));
  fill_f32<<<1, 256, 0, stream>>>((float*)zeros, 0.0f, DDE / 2);

  {  // reverse-edge lookup + attention mask (layer-invariant)
    size_t nn = (size_t)GN * GN;
    fill_i32<<<dim3((unsigned)((nn + 255) / 256)), 256, 0, stream>>>(table, -1, nn);
    mask_init<<<dim3((unsigned)((nn + 255) / 256)), 256, 0, stream>>>(mask);
    rev_set<<<GE / 256, 256, 0, stream>>>(table, srcI, dstI);
    rev_get<<<GE / 256, 256, 0, stream>>>(table, srcI, dstI, rev);
    mask_edges<<<GE / 256, 256, 0, stream>>>(mask, srcI, dstI, pos);
  }

  const float bn_scale = 1.0f / sqrtf(1.0f + 1e-5f);
  GatherArgs ga0{};
  ConcatArgs ca0{};

  const u16* nf_cur = nfA_bf;
  const u16* ef_cur = efA_bf;
  for (int l = 0; l < NLY; ++l) {
    const int last = (l == NLY - 1);
    const int last_relu = last ? 0 : 1;
    void* nf_next = last ? (void*)d_out : (void*)nfB_bf;
    void* ef_next = last ? (void*)((float*)d_out + (size_t)GN * DDN) : (void*)efB_bf;
    const u16* inw_l = inw_bf + (size_t)l * 3 * DDN * DDN;
    const float* inb_l = inb + (size_t)l * 3 * DDN;
    const u16* ow_l = ow_bf + (size_t)l * DDN * DDN;
    const float* ob_l = ob + (size_t)l * DDN;
    const u16* etw_l = etw_bf + (size_t)l * DDE * DDE;
    const float* etb_l = etb + (size_t)l * DDE;
    const float* etg_l = etg + (size_t)l * DDE;
    const float* etbb_l = etbb + (size_t)l * DDE;
    const u16* unw_l = unw_bf + (size_t)l * DDN * (DDN + DDE);
    const float* unb_l = unb + (size_t)l * DDN;
    const u16* uew_l = uew_bf + (size_t)l * DDE * (2 * DDN + 2 * DDE);
    const float* ueb_l = ueb + (size_t)l * DDE;

    // 1) qkv = nf @ inw^T + inb  -> bf16 [2048 x 1536]
    {
      GemmEpi ep{inb_l, nullptr, nullptr, nullptr, 1.f, 1.f, 0, 0, 0};
      gemm_kernel<0, 0><<<dim3(1536 / 64, GN / 128, 1), 256, 0, stream>>>(
          GN, 1536, DDN, nf_cur, DDN, 0, inw_l, DDN, 0, qkv_bf, 1536, 0, 1, ga0, ca0, ep);
    }
    // 2) scores[h] = (q k^T)/8 + mask -> f32, per head [2048 x 2048], K=64
    {
      GemmEpi ep{nullptr, nullptr, nullptr, mask, 0.125f, 1.f, 0, 3, GN};
      gemm_kernel<0, 0><<<dim3(GN / 64, GN / 128, NHD), 256, 0, stream>>>(
          GN, GN, HDM, qkv_bf, 1536, HDM, qkv_bf + DDN, 1536, HDM,
          scores, GN, (long long)GN * GN, 0, ga0, ca0, ep);
    }
    // 3) softmax (f32) -> bf16 probabilities
    softmax_rows<<<NHD * GN, 256, 0, stream>>>(scores, probs, GN);
    // 4) ctx[h] = P @ V -> bf16 [2048 x 64], K=2048 (B K-major)
    {
      GemmEpi ep{nullptr, nullptr, nullptr, nullptr, 1.f, 1.f, 0, 0, 0};
      gemm_kernel<0, 1><<<dim3(1, GN / 128, NHD), 256, 0, stream>>>(
          GN, HDM, GN, probs, GN, (long long)GN * GN, qkv_bf + 2 * DDN, 1536, HDM,
          ctx_bf, DDN, HDM, 1, ga0, ca0, ep);
    }
    // 5) x_upd = ctx @ ow^T + ob -> bf16
    {
      GemmEpi ep{ob_l, nullptr, nullptr, nullptr, 1.f, 1.f, 0, 0, 0};
      gemm_kernel<0, 0><<<dim3(DDN / 64, GN / 128, 1), 256, 0, stream>>>(
          GN, DDN, DDN, ctx_bf, DDN, 0, ow_l, DDN, 0, xupd_bf, DDN, 0, 1, ga0, ca0, ep);
    }
    // 6) te = relu((ef @ etw^T + etb)*bn_scale*etg + etbb) -> f32 [32768 x 512]
    {
      GemmEpi ep{etb_l, etg_l, etbb_l, nullptr, 1.f, bn_scale, 1, 2, 0};
      gemm_kernel<0, 0><<<dim3(DDE / 64, GE / 128, 1), 256, 0, stream>>>(
          GE, DDE, DDE, ef_cur, DDE, 0, etw_l, DDE, 0, te, DDE, 0, 0, ga0, ca0, ep);
    }
    // 7) segment-max to subj/obj; fuse subj+obj -> bf16
    fill_f32<<<(GN * DDE) / 256, 256, 0, stream>>>(subj, 0.0f, (size_t)GN * DDE);
    fill_f32<<<(GN * DDE) / 256, 256, 0, stream>>>(obj, 0.0f, (size_t)GN * DDE);
    scatter_max<<<dim3((unsigned)(((size_t)GE * DDE) / 256)), 256, 0, stream>>>(
        te, srcI, dstI, subj, obj);
    sum_cvt_bf16<<<(GN * DDE) / 8 / 256, 256, 0, stream>>>(subj, obj, s1_bf, (size_t)GN * DDE);
    // 8) ef_next = [nf[src]|ef|ef[rev]|nf[dst]] @ uew^T + ueb   [32768 x 512], K=2048
    {
      GatherArgs ga{srcI, dstI, rev, nf_cur, ef_cur, zeros};
      GemmEpi ep{ueb_l, nullptr, nullptr, nullptr, 1.f, 1.f, last_relu, 0, 0};
      gemm_kernel<1, 0><<<dim3(DDE / 64, GE / 128, 1), 256, 0, stream>>>(
          GE, DDE, 2 * DDN + 2 * DDE, nullptr, 0, 0, uew_l, 2 * DDN + 2 * DDE, 0,
          ef_next, DDE, 0, last ? 0 : 1, ga, ca0, ep);
    }
    // 9) nf_next = [x_upd | subj+obj] @ unw^T + unb  [2048 x 512], K=1024
    {
      ConcatArgs ca{xupd_bf, s1_bf};
      GemmEpi ep{unb_l, nullptr, nullptr, nullptr, 1.f, 1.f, last_relu, 0, 0};
      gemm_kernel<2, 0><<<dim3(DDN / 64, GN / 128, 1), 256, 0, stream>>>(
          GN, DDN, DDN + DDE, nullptr, 0, 0, unw_l, DDN + DDE, 0,
          nf_next, DDN, 0, last ? 0 : 1, ga0, ca, ep);
    }
    nf_cur = nfB_bf;
    ef_cur = efB_bf;
  }
}
